// IDGCN_24481313587440
// MI455X (gfx1250) — matrix operand order repaired and checked by reading, not compiled, not run on hardware
//
#include <hip/hip_runtime.h>

typedef __attribute__((ext_vector_type(16))) __bf16 v16bf;
typedef __attribute__((ext_vector_type(8)))  float  v8f;

#define Bb 32
#define Ss 512
#define Ee 768
#define Hh 300
#define HhP 320      // K-padded recurrent dim
#define H4 1200
#define H4P 1280     // N-padded gate dim (multiple of 128)
#define F2 600
#define F2P 640      // padded GAT feature dim (multiple of 128 and 32)
#define Pp 5
#define BS (Bb*Ss)
#define NEGBIG (-9.0e15f)

// ---------------- staging: fp32 [inR,inC] -> zero-padded bf16 [outR,outC] ----------------
__global__ void k_stage_pad_bf16(const float* __restrict__ in, __bf16* __restrict__ out,
                                 int inR, int inC, int outR, int outC) {
  int total = outR * outC;
  for (int i = blockIdx.x * blockDim.x + threadIdx.x; i < total; i += gridDim.x * blockDim.x) {
    int r = i / outC, c = i - r * outC;
    out[i] = (r < inR && c < inC) ? (__bf16)in[(size_t)r * inC + c] : (__bf16)0.0f;
  }
}

__global__ void k_pad_f32(const float* __restrict__ in, float* __restrict__ out, int n, int np) {
  int i = blockIdx.x * blockDim.x + threadIdx.x;
  if (i >= np) return;
  out[i] = (i < n) ? in[i] : 0.f;
}

// rowmap for the reverse-padded backward-LSTM input (fold reversal into GEMM A-fetch)
__global__ void k_rowmap_rev(const int* __restrict__ len, int* __restrict__ rowmap) {
  int i = blockIdx.x * blockDim.x + threadIdx.x;
  if (i >= BS) return;
  int b = i / Ss, t = i % Ss;
  int L = len[b];
  rowmap[i] = (t < L) ? (b * Ss + (L - 1 - t)) : -1;
}

// ---------------- guard-free bf16 WMMA GEMM:  C[M,N] = A[rowmap][K] @ W[K,N] (+bias) ----------------
// Preconditions (enforced by padded staging): K%32==0, N%128==0, 16B-aligned rows.
// 256 threads = 8 waves; block tile 64x128; BK=32; each wave: 16 rows x 64 cols (4 wmma tiles).
__global__ void __launch_bounds__(256) k_gemm_bf16(
    const __bf16* __restrict__ A, int lda,
    const __bf16* __restrict__ W,
    const int* __restrict__ rowmap,
    const float* __restrict__ bias,
    float* __restrict__ Cf, __bf16* __restrict__ Cb,
    int M, int N, int K)
{
  __shared__ __bf16 As[64 * 40];    // padded stride 40 halves (80 B, 16-mult)
  __shared__ __bf16 Bs[32 * 136];   // padded stride 136 halves (272 B, 16-mult)

  const int tid   = threadIdx.x;
  const int lane  = tid & 31;
  const int wid   = tid >> 5;
  const int strip = wid & 3;        // row strip (16 rows)
  const int chalf = wid >> 2;       // 0/1: which 64-col half
  const int m0 = blockIdx.y * 64;
  const int n0 = blockIdx.x * 128;

  const v8f vzero = {0.f,0.f,0.f,0.f,0.f,0.f,0.f,0.f};
  v8f acc[4];
#pragma unroll
  for (int ct = 0; ct < 4; ++ct) acc[ct] = vzero;

  // cooperative-load index split
  const int ar  = tid >> 2;          // 0..63 (A row)
  const int acs = (tid & 3) * 8;     // A col segment (8 halves = 16 B)
  int asrc = m0 + ar;
  if (rowmap) asrc = rowmap[asrc];
  const int bkr = tid >> 3;          // 0..31 (B row)
  const int bcs = (tid & 7) * 16;    // B col segment (16 halves = 32 B)

  const uint4 z4 = {0u, 0u, 0u, 0u};

  for (int k0 = 0; k0 < K; k0 += 32) {
    // A tile: one 16B vector load per thread (zero row for padded reverse map)
    uint4 va = z4;
    if (asrc >= 0) va = *(const uint4*)(A + (size_t)asrc * lda + k0 + acs);
    *(uint4*)(As + ar * 40 + acs) = va;
    // B tile: two 16B vector loads per thread
    const uint4* wp = (const uint4*)(W + (size_t)(k0 + bkr) * N + n0 + bcs);
    uint4 w0 = wp[0];
    uint4 w1 = wp[1];
    *(uint4*)(Bs + bkr * 136 + bcs)     = w0;
    *(uint4*)(Bs + bkr * 136 + bcs + 8) = w1;
    if (k0 + 32 < K) {  // global_prefetch_b8 for next tiles
      __builtin_prefetch(W + (size_t)(k0 + 32 + bkr) * N + n0 + bcs, 0, 1);
      if (asrc >= 0) __builtin_prefetch(A + (size_t)asrc * lda + k0 + 32 + acs, 0, 1);
    }
    __syncthreads();

    // A fragment (16x32 bf16, ISA layout): lane l -> row l&15, K pairs per (l>>4)
    union { unsigned u[8]; v16bf v; } ua;
    {
      const int row = strip * 16 + (lane & 15);
      const int hi = lane >> 4;
      const unsigned* arow = (const unsigned*)(As + row * 40);
#pragma unroll
      for (int v = 0; v < 4; ++v) {
        ua.u[v]     = arow[hi * 4 + v];
        ua.u[4 + v] = arow[8 + hi * 4 + v];
      }
    }
#pragma unroll
    for (int ct = 0; ct < 4; ++ct) {
      // B fragment (32x16 bf16): lane l -> N=l&15, K in table order per (l>>4)
      union { __bf16 h[16]; v16bf v; } ub;
      const __bf16* bcol = Bs + chalf * 64 + ct * 16 + (lane & 15);
      const int bhi = lane >> 4;
#pragma unroll
      for (int j = 0; j < 16; ++j) ub.h[j] = bcol[((j < 8) ? (8 * bhi + j) : (16 + 8 * bhi + j - 8)) * 136];
      acc[ct] = __builtin_amdgcn_wmma_f32_16x16x32_bf16(
          false, ua.v, false, ub.v, (short)0, acc[ct], false, false);
    }
    __syncthreads();
  }

  // epilogue: C layout lane=N, vgpr r -> M=r (lanes>=16: M=r+8); N padded -> no store guard
#pragma unroll
  for (int ct = 0; ct < 4; ++ct) {
    int n = n0 + chalf * 64 + ct * 16 + (lane & 15);
    float bv = bias ? bias[n] : 0.0f;
#pragma unroll
    for (int r = 0; r < 8; ++r) {
      int m = m0 + strip * 16 + ((lane < 16) ? r : r + 8);
      float v = acc[ct][r] + bv;
      if (Cf) Cf[(size_t)m * N + n] = v;
      else    Cb[(size_t)m * N + n] = (__bf16)v;
    }
  }
}

// ---------------- persistent BiLSTM recurrence ----------------
// grid = 2 (dir 0 fwd, dir 1 bwd-on-reversed-input); 256 thr = 8 waves.
// Per step: G[32,1200] = gates_in(t) + Hbf[32,320] @ Wh[320,1200] via WMMA, then gate math.
// Wh rows 300..319 are staged zero -> no K guard; hbf cols 300..327 stay zero.
__global__ void __launch_bounds__(256) k_lstm_scan(
    const float* __restrict__ gates_all,   // [2][BS*H4P] (x@Wi + b precomputed, padded stride)
    const __bf16* __restrict__ Wh_all,     // [2][HhP*H4] bf16, zero-padded rows
    const int* __restrict__ len,
    float* __restrict__ out_all)           // [2][BS*Hh]
{
  extern __shared__ char smem[];
  float*  G   = (float*)smem;                                   // 32*1200 f32
  float*  hS  = (float*)(smem + 32 * H4 * 4);                   // 32*300 f32
  float*  cS  = (float*)(smem + 32 * H4 * 4 + 32 * Hh * 4);     // 32*300 f32
  __bf16* hbf = (__bf16*)(smem + 32 * H4 * 4 + 2 * 32 * Hh * 4);// 32*328 bf16

  const int dir = blockIdx.x;
  const float*  gates = gates_all + (size_t)dir * BS * H4P;
  const __bf16* Wh    = Wh_all   + (size_t)dir * HhP * H4;
  float* out = out_all + (size_t)dir * BS * Hh;

  const int tid  = threadIdx.x;
  const int lane = tid & 31;
  const int wid  = tid >> 5;
  const v8f vzero = {0.f,0.f,0.f,0.f,0.f,0.f,0.f,0.f};

  for (int j = tid; j < 32 * Hh; j += 256) { hS[j] = 0.f; cS[j] = 0.f; }
  for (int j = tid; j < 32 * 328; j += 256) hbf[j] = (__bf16)0.0f;
  __syncthreads();

  for (int t = 0; t < Ss; ++t) {
    // phase 1: G <- precomputed x@Wi + b at timestep t (float4 staging)
    for (int j = tid; j < 32 * 300; j += 256) {   // 32 rows x 300 float4 = 1200 cols
      int b = j / 300, n4 = j - b * 300;
      ((float4*)G)[b * 300 + n4] =
          ((const float4*)(gates + ((size_t)(b * Ss + t)) * H4P))[n4];
    }
    __syncthreads();

    // phase 2: G += Hbf @ Wh (wave-parallel over 75 column tiles, 2 row strips per wave)
    for (int ct = wid; ct < 75; ct += 8) {
      v8f a0 = vzero, a1 = vzero;
      for (int k0 = 0; k0 < HhP; k0 += 32) {
        // B fragment straight from L2-resident Wh: lane -> N column l&15, K in table order per (l>>4)
        union { __bf16 h[16]; v16bf v; } ub;
        const __bf16* bp = Wh + (size_t)k0 * H4 + ct * 16 + (lane & 15);
        const int bhi = lane >> 4;
#pragma unroll
        for (int j = 0; j < 16; ++j) ub.h[j] = bp[(size_t)((j < 8) ? (8 * bhi + j) : (16 + 8 * bhi + j - 8)) * H4];
        union { unsigned u[8]; v16bf v; } ua0, ua1;
        const int hi = lane >> 4;
        const unsigned* h0p = (const unsigned*)(hbf + (lane & 15) * 328);
        const unsigned* h1p = (const unsigned*)(hbf + (16 + (lane & 15)) * 328);
        const int kb = k0 >> 1;
#pragma unroll
        for (int v = 0; v < 4; ++v) {
          ua0.u[v]     = h0p[kb + hi * 4 + v];
          ua0.u[4 + v] = h0p[kb + 8 + hi * 4 + v];
          ua1.u[v]     = h1p[kb + hi * 4 + v];
          ua1.u[4 + v] = h1p[kb + 8 + hi * 4 + v];
        }
        a0 = __builtin_amdgcn_wmma_f32_16x16x32_bf16(false, ua0.v, false, ub.v, (short)0, a0, false, false);
        a1 = __builtin_amdgcn_wmma_f32_16x16x32_bf16(false, ua1.v, false, ub.v, (short)0, a1, false, false);
      }
      int n = ct * 16 + (lane & 15);
#pragma unroll
      for (int r = 0; r < 8; ++r) {
        int m = (lane < 16) ? r : (r + 8);
        G[m * H4 + n]        += a0[r];
        G[(m + 16) * H4 + n] += a1[r];
      }
    }
    __syncthreads();

    // phase 3: gate nonlinearity + masked state update (torch gate order i,f,g,o)
    for (int j = tid; j < 32 * Hh; j += 256) {
      int b = j / Hh, u = j - b * Hh;
      float gi = G[b * H4 + u];
      float gf = G[b * H4 + 300 + u];
      float gg = G[b * H4 + 600 + u];
      float go = G[b * H4 + 900 + u];
      float si = 1.f / (1.f + __expf(-gi));
      float sf = 1.f / (1.f + __expf(-gf));
      float tg = tanhf(gg);
      float so = 1.f / (1.f + __expf(-go));
      float cn = sf * cS[j] + si * tg;
      float hn = so * tanhf(cn);
      bool valid = t < len[b];
      float hcur = valid ? hn : hS[j];
      float ccur = valid ? cn : cS[j];
      hS[j] = hcur; cS[j] = ccur;
      hbf[b * 328 + u] = (__bf16)hcur;
      out[((size_t)(b * Ss + t)) * Hh + u] = valid ? hn : 0.f;
    }
    __syncthreads();
  }
}

// text_out[B*S,600] (fp32) = concat(out_f, reverse_padded(out_b)); bf16 copy padded to stride 640
__global__ void k_assemble(const float* __restrict__ out_f, const float* __restrict__ out_b,
                           const int* __restrict__ len,
                           float* __restrict__ text_out, __bf16* __restrict__ text_out_bf) {
  int i = blockIdx.x * blockDim.x + threadIdx.x;
  if (i >= BS * F2P) return;
  int f = i % F2P, row = i / F2P;
  if (f >= F2) { text_out_bf[i] = (__bf16)0.0f; return; }
  int b = row / Ss, t = row % Ss;
  float v;
  if (f < Hh) {
    v = out_f[(size_t)row * Hh + f];
  } else {
    int L = len[b];
    v = (t < L) ? out_b[((size_t)(b * Ss + (L - 1 - t))) * Hh + (f - Hh)] : 0.f;
  }
  text_out[(size_t)row * F2 + f] = v;
  text_out_bf[i] = (__bf16)v;
}

// e1/e2 attention logit vectors for head 0 (Wh0 stride F2P)
__global__ void k_evec(const __bf16* __restrict__ Wh0, const float* __restrict__ a1,
                       const float* __restrict__ a2, float* __restrict__ e1, float* __restrict__ e2) {
  int row = blockIdx.x * blockDim.x + threadIdx.x;
  if (row >= BS) return;
  const __bf16* w = Wh0 + (size_t)row * F2P;
  float s1 = 0.f, s2 = 0.f;
  for (int f = 0; f < F2; ++f) { float v = (float)w[f]; s1 += v * a1[f]; s2 += v * a2[f]; }
  e1[row] = s1; e2[row] = s2;
}

// only head-0 GAT rows at contra_pos survive the mask: softmax row + att@Wh + relu(elu)
// then per-row contribution x_row . text_out[t] -> partial (deterministic reduce later)
__global__ void __launch_bounds__(256) k_gat_rows(
    const int* __restrict__ contra, const int* __restrict__ adj,
    const float* __restrict__ e1, const float* __restrict__ e2,
    const __bf16* __restrict__ Wh0, const float* __restrict__ text_out,
    float* __restrict__ partial)
{
  const int b = blockIdx.x / Pp, p = blockIdx.x % Pp;
  const int tid = threadIdx.x;
  const int s = contra[b * Pp + p];
  bool dup = false;
  for (int q = 0; q < p; ++q) if (contra[b * Pp + q] == s) dup = true;
  float* prow = partial + (size_t)(b * Pp + p) * Ss;
  if (dup) {  // mask .set(1.0) -> duplicates contribute once
    for (int t = tid; t < Ss; t += 256) prow[t] = 0.f;
    return;
  }
  __shared__ float att[Ss];
  __shared__ float red[256];
  __shared__ float xrow[F2];
  const float e1v = e1[b * Ss + s];
  for (int t = tid; t < Ss; t += 256) {
    float e = e1v + e2[b * Ss + t];
    float lr = e > 0.f ? e : 0.2f * e;
    att[t] = (adj[((size_t)(b * Ss + s)) * Ss + t] > 0) ? lr : NEGBIG;
  }
  __syncthreads();
  red[tid] = fmaxf(att[tid], att[tid + 256]); __syncthreads();
  for (int st = 128; st > 0; st >>= 1) { if (tid < st) red[tid] = fmaxf(red[tid], red[tid + st]); __syncthreads(); }
  const float mx = red[0]; __syncthreads();
  float ls = 0.f;
  for (int t = tid; t < Ss; t += 256) { float ev = __expf(att[t] - mx); att[t] = ev; ls += ev; }
  red[tid] = ls; __syncthreads();
  for (int st = 128; st > 0; st >>= 1) { if (tid < st) red[tid] += red[tid + st]; __syncthreads(); }
  const float inv = 1.f / red[0]; __syncthreads();
  for (int t = tid; t < Ss; t += 256) att[t] *= inv;
  __syncthreads();
  for (int c = tid; c < F2; c += 256) {
    float acc = 0.f;
    for (int t = 0; t < Ss; ++t) acc += att[t] * (float)Wh0[((size_t)(b * Ss + t)) * F2P + c];
    xrow[c] = acc > 0.f ? acc : 0.f;  // relu(elu(x)) == max(x,0)
  }
  __syncthreads();
  for (int t = tid; t < Ss; t += 256) {
    const float4* to = (const float4*)(text_out + ((size_t)(b * Ss + t)) * F2);
    const float4* xr = (const float4*)xrow;
    float acc = 0.f;
    for (int f = 0; f < F2 / 4; ++f) {
      float4 a = to[f], x = xr[f];
      acc += a.x * x.x + a.y * x.y + a.z * x.z + a.w * x.w;
    }
    prow[t] = acc;
  }
}

// alpha = softmax(sum_p partial); pooled = alpha @ text_out
__global__ void __launch_bounds__(256) k_alpha_pool(
    const float* __restrict__ partial, const float* __restrict__ text_out,
    float* __restrict__ pooled)
{
  const int b = blockIdx.x, tid = threadIdx.x;
  __shared__ float al[Ss];
  __shared__ float red[256];
  for (int t = tid; t < Ss; t += 256) {
    float s = 0.f;
    for (int p = 0; p < Pp; ++p) s += partial[(size_t)(b * Pp + p) * Ss + t];
    al[t] = s;
  }
  __syncthreads();
  red[tid] = fmaxf(al[tid], al[tid + 256]); __syncthreads();
  for (int st = 128; st > 0; st >>= 1) { if (tid < st) red[tid] = fmaxf(red[tid], red[tid + st]); __syncthreads(); }
  const float mx = red[0]; __syncthreads();
  float ls = 0.f;
  for (int t = tid; t < Ss; t += 256) { float ev = __expf(al[t] - mx); al[t] = ev; ls += ev; }
  red[tid] = ls; __syncthreads();
  for (int st = 128; st > 0; st >>= 1) { if (tid < st) red[tid] += red[tid + st]; __syncthreads(); }
  const float inv = 1.f / red[0]; __syncthreads();
  for (int t = tid; t < Ss; t += 256) al[t] *= inv;
  __syncthreads();
  for (int f = tid; f < F2; f += 256) {
    float acc = 0.f;
    for (int t = 0; t < Ss; ++t) acc += al[t] * text_out[((size_t)(b * Ss + t)) * F2 + f];
    pooled[b * F2 + f] = acc;
  }
}

// classifier: h1 = pooled@fc1_W + b1 ; out = h1@fc_W + b
__global__ void __launch_bounds__(512) k_fc(
    const float* __restrict__ pooled,
    const float* __restrict__ fc1W, const float* __restrict__ fc1b,
    const float* __restrict__ fcW,  const float* __restrict__ fcb,
    float* __restrict__ out)
{
  const int b = blockIdx.x, j = threadIdx.x;
  __shared__ float h1[512];
  float acc = fc1b[j];
  for (int k = 0; k < F2; ++k) acc += pooled[b * F2 + k] * fc1W[k * 512 + j];
  h1[j] = acc;
  __syncthreads();
  if (j < 3) {
    float o = fcb[j];
    for (int q = 0; q < 512; ++q) o += h1[q] * fcW[q * 3 + j];
    out[b * 3 + j] = o;
  }
}

extern "C" void kernel_launch(void* const* d_in, const int* in_sizes, int n_in,
                              void* d_out, int out_size, void* d_ws, size_t ws_size,
                              hipStream_t stream) {
  (void)in_sizes; (void)n_in; (void)out_size; (void)ws_size;
  const float* text_bert = (const float*)d_in[0];
  const int*   text_len  = (const int*)d_in[1];
  const int*   contra    = (const int*)d_in[2];
  const int*   adj       = (const int*)d_in[3];
  const float* Wi_f = (const float*)d_in[4];
  const float* Wh_f = (const float*)d_in[5];
  const float* b_f  = (const float*)d_in[6];
  const float* Wi_b = (const float*)d_in[7];
  const float* Wh_b = (const float*)d_in[8];
  const float* b_b  = (const float*)d_in[9];
  const float* gat_W  = (const float*)d_in[10];
  const float* gat_a1 = (const float*)d_in[11];
  const float* gat_a2 = (const float*)d_in[12];
  const float* fc1W = (const float*)d_in[13];
  const float* fc1b = (const float*)d_in[14];
  const float* fcW  = (const float*)d_in[15];
  const float* fcb  = (const float*)d_in[16];

  char* ws = (char*)d_ws;
  auto alloc = [&](size_t bytes) { char* p = ws; ws += (bytes + 255) & ~(size_t)255; return p; };
  __bf16* Abf    = (__bf16*)alloc((size_t)BS * Ee * 2);            // [16384,768]
  __bf16* Wi_bf  = (__bf16*)alloc((size_t)2 * Ee * H4P * 2);       // [768,1280] x2, zero-padded cols
  __bf16* Wh_bf  = (__bf16*)alloc((size_t)2 * HhP * H4 * 2);       // [320,1200] x2, zero-padded rows
  __bf16* gW0bf  = (__bf16*)alloc((size_t)F2P * F2P * 2);          // [640,640], zero-padded
  float*  biasP  = (float*)alloc((size_t)2 * H4P * 4);             // padded biases
  float*  gates  = (float*)alloc((size_t)2 * BS * H4P * 4);        // stride 1280
  float*  outfb  = (float*)alloc((size_t)2 * BS * Hh * 4);
  float*  text_out = (float*)alloc((size_t)BS * F2 * 4);
  __bf16* text_out_bf = (__bf16*)alloc((size_t)BS * F2P * 2);      // stride 640
  __bf16* Wh0bf  = (__bf16*)alloc((size_t)BS * F2P * 2);           // stride 640
  float*  e1     = (float*)alloc((size_t)BS * 4);
  float*  e2     = (float*)alloc((size_t)BS * 4);
  int*    rowmap = (int*)alloc((size_t)BS * 4);
  float*  partial= (float*)alloc((size_t)Bb * Pp * Ss * 4);
  float*  pooled = (float*)alloc((size_t)Bb * F2 * 4);

  // bf16 staging (zero-padded so the GEMM/scan have no bounds guards)
  k_stage_pad_bf16<<<2048, 256, 0, stream>>>(text_bert, Abf, BS, Ee, BS, Ee);
  k_stage_pad_bf16<<<960, 256, 0, stream>>>(Wi_f, Wi_bf, Ee, H4, Ee, H4P);
  k_stage_pad_bf16<<<960, 256, 0, stream>>>(Wi_b, Wi_bf + (size_t)Ee * H4P, Ee, H4, Ee, H4P);
  k_stage_pad_bf16<<<512, 256, 0, stream>>>(Wh_f, Wh_bf, Hh, H4, HhP, H4);
  k_stage_pad_bf16<<<512, 256, 0, stream>>>(Wh_b, Wh_bf + (size_t)HhP * H4, Hh, H4, HhP, H4);
  k_stage_pad_bf16<<<512, 256, 0, stream>>>(gat_W, gW0bf, F2, F2, F2P, F2P);  // head 0 only
  k_pad_f32<<<(2 * H4P + 255) / 256, 256, 0, stream>>>(b_f, biasP, H4, H4P);
  k_pad_f32<<<(2 * H4P + 255) / 256, 256, 0, stream>>>(b_b, biasP + H4P, H4, H4P);
  k_rowmap_rev<<<(BS + 255) / 256, 256, 0, stream>>>(text_len, rowmap);

  // gate pre-activations (x @ Wi + b) for both directions: N=1280, K=768 (guard-free)
  dim3 g1(H4P / 128, BS / 64);
  k_gemm_bf16<<<g1, 256, 0, stream>>>(Abf, Ee, Wi_bf, nullptr, biasP,
                                      gates, nullptr, BS, H4P, Ee);
  k_gemm_bf16<<<g1, 256, 0, stream>>>(Abf, Ee, Wi_bf + (size_t)Ee * H4P, rowmap, biasP + H4P,
                                      gates + (size_t)BS * H4P, nullptr, BS, H4P, Ee);

  // sequential recurrence: both directions concurrently, WMMA per step, state in LDS
  size_t smem = (size_t)32 * H4 * 4 + 2 * (size_t)32 * Hh * 4 + (size_t)32 * 328 * 2; // 251,392 B
  (void)hipFuncSetAttribute((const void*)k_lstm_scan,
                            hipFuncAttributeMaxDynamicSharedMemorySize, (int)smem);
  k_lstm_scan<<<2, 256, smem, stream>>>(gates, Wh_bf, text_len, outfb);

  k_assemble<<<(BS * F2P + 255) / 256, 256, 0, stream>>>(
      outfb, outfb + (size_t)BS * Hh, text_len, text_out, text_out_bf);

  // head-0 GAT projection: N=640, K=640 (guard-free)
  dim3 g2(F2P / 128, BS / 64);
  k_gemm_bf16<<<g2, 256, 0, stream>>>(text_out_bf, F2P, gW0bf, nullptr, nullptr,
                                      nullptr, Wh0bf, BS, F2P, F2P);

  k_evec<<<(BS + 255) / 256, 256, 0, stream>>>(Wh0bf, gat_a1, gat_a2, e1, e2);
  k_gat_rows<<<Bb * Pp, 256, 0, stream>>>(contra, adj, e1, e2, Wh0bf, text_out, partial);
  k_alpha_pool<<<Bb, 256, 0, stream>>>(partial, text_out, pooled);
  k_fc<<<Bb, 512, 0, stream>>>(pooled, fc1W, fc1b, fcW, fcb, (float*)d_out);
}
